// QuantizedConv1D_41154376630556
// MI455X (gfx1250) — matrix operand order repaired and checked by reading, not compiled, not run on hardware
//
#include <hip/hip_runtime.h>
#include <stdint.h>
#include <string.h>

// ---------------------------------------------------------------------------
// out[M,N] = x[M,K] @ (int8_w[K,N] * scale) + bias[N]
//   M = 32768, K = 1024, N = 4096
//
// MI455X plan:
//  - pre-pass: w -> bf16 transposed [N][K] (8 MB), x -> bf16 [M][K] (64 MB)
//    in d_ws (int8 exact in bf16; scale folded into fp32 epilogue)
//  - main GEMM: 256x128x32 block tile, 8 wave32 waves (64x64 wave tile,
//    16 v_wmma_f32_16x16x32_bf16 per wave per k-step), double-buffered
//    GLOBAL_LOAD_ASYNC_TO_LDS_B128 staging (ASYNCcnt), one barrier/iter.
// ---------------------------------------------------------------------------

typedef __bf16        v16bf  __attribute__((ext_vector_type(16)));
typedef float         v8f    __attribute__((ext_vector_type(8)));
typedef float         f32x4  __attribute__((ext_vector_type(4)));
typedef unsigned int  u32x4  __attribute__((ext_vector_type(4)));
typedef unsigned int  u32x2  __attribute__((ext_vector_type(2)));

static constexpr int K_DIM = 1024;      // NX
static constexpr int N_DIM = 4096;      // NF
static constexpr int M_DIM = 8 * 4096;  // B*S

union FragU {
    struct { u32x4 lo; u32x4 hi; } u;
    v16bf v;
};

__device__ __forceinline__ unsigned short f2bf(float f) {
    unsigned u = __builtin_bit_cast(unsigned, f);
    u += 0x7FFFu + ((u >> 16) & 1u);
    return (unsigned short)(u >> 16);
}
__device__ __forceinline__ unsigned cvt2(float a, float b) {
#if __has_builtin(__builtin_amdgcn_cvt_pk_bf16_f32)
    typedef __bf16 v2bf __attribute__((ext_vector_type(2)));
    v2bf r = __builtin_amdgcn_cvt_pk_bf16_f32(a, b);
    return __builtin_bit_cast(unsigned, r);
#else
    return (unsigned)f2bf(a) | ((unsigned)f2bf(b) << 16);
#endif
}

// --- gfx1250 async copy global->LDS (ASYNCcnt), via inline asm -------------
__device__ __forceinline__ void async_ld16(const void* gaddr, unsigned lds_off) {
    asm volatile("global_load_async_to_lds_b128 %0, %1, off"
                 :: "v"(lds_off), "v"(gaddr) : "memory");
}
__device__ __forceinline__ void wait_async0() {
    asm volatile("s_wait_asynccnt 0x0" ::: "memory");
}

// ============================ pre-pass kernels =============================

extern "C" __global__ __launch_bounds__(256)
void convert_w_kernel(const int8_t* __restrict__ w,
                      unsigned short* __restrict__ wT)
{
    const int n = blockIdx.x * 256 + threadIdx.x;
    for (int k0 = 0; k0 < K_DIM; k0 += 8) {
        float f[8];
#pragma unroll
        for (int i = 0; i < 8; ++i)
            f[i] = (float)w[(size_t)(k0 + i) * N_DIM + n];
        u32x4 p = { cvt2(f[0], f[1]), cvt2(f[2], f[3]),
                    cvt2(f[4], f[5]), cvt2(f[6], f[7]) };
        *(u32x4*)&wT[(size_t)n * K_DIM + k0] = p;
    }
}

extern "C" __global__ __launch_bounds__(256)
void convert_x_kernel(const float* __restrict__ x,
                      unsigned short* __restrict__ xb)
{
    const size_t i = ((size_t)blockIdx.x * 256 + threadIdx.x) * 8;
    f32x4 v0 = *(const f32x4*)(x + i);
    f32x4 v1 = *(const f32x4*)(x + i + 4);
    u32x4 p = { cvt2(v0.x, v0.y), cvt2(v0.z, v0.w),
                cvt2(v1.x, v1.y), cvt2(v1.z, v1.w) };
    *(u32x4*)&xb[i] = p;
}

// ==================== fast GEMM: async-LDS double buffer ===================

static constexpr int ABM = 256, ABN = 128, ABK = 32;
static constexpr int A_BYTES = ABM * ABK * 2;       // 16 KB
static constexpr int B_BYTES = ABN * ABK * 2;       //  8 KB
static constexpr int STG     = A_BYTES + B_BYTES;   // 24 KB
static constexpr int DEPTH   = 2;                   // 48 KB LDS total

extern "C" __global__ __launch_bounds__(256)
void qlinear_gemm_async(const unsigned short* __restrict__ xb,   // bf16 [M][K]
                        const unsigned short* __restrict__ wT,   // bf16 [N][K]
                        const float* __restrict__ wscale,
                        const float* __restrict__ bias,
                        float* __restrict__ out)
{
    __shared__ __align__(16) char lds[DEPTH * STG];

    const int tid  = threadIdx.x;
    const int lane = tid & 31;
    const int wid  = tid >> 5;
    const int wm   = wid & 3;   // 4 waves along M: 64 rows each
    const int wn   = wid >> 2;  // 2 waves along N: 64 cols each

    const int m0 = blockIdx.y * ABM;
    const int n0 = blockIdx.x * ABN;

    const unsigned lds_base = (unsigned)(size_t)&lds[0];
    const unsigned t16 = (unsigned)tid * 16u;

    const char* agbase = (const char*)(xb + (size_t)m0 * K_DIM);
    const char* bgbase = (const char*)(wT + (size_t)n0 * K_DIM);

    // stage one 256x32 A tile (16KB) + 128x32 B tile (8KB) into buffer kt&1.
    // each async op: 4 lanes x 16B cover one 64B row segment; rows 2KB apart.
    auto stage = [&](int kt) {
        const unsigned sbase = lds_base + (unsigned)((kt & 1) * STG);
        const size_t   kbyte = (size_t)(kt * ABK) * 2;   // 64B per row seg
#pragma unroll
        for (int s = 0; s < 4; ++s) {                    // A: 4 sweeps
            unsigned off = (unsigned)s * 4096u + t16;
            unsigned row = off >> 6, col = off & 63u;
            async_ld16(agbase + (size_t)row * (K_DIM * 2) + kbyte + col,
                       sbase + off);
        }
#pragma unroll
        for (int s = 0; s < 2; ++s) {                    // B: 2 sweeps
            unsigned off = (unsigned)s * 4096u + t16;
            unsigned row = off >> 6, col = off & 63u;
            async_ld16(bgbase + (size_t)row * (K_DIM * 2) + kbyte + col,
                       sbase + A_BYTES + off);
        }
    };

    v8f acc[4][4];
#pragma unroll
    for (int i = 0; i < 4; ++i)
#pragma unroll
        for (int j = 0; j < 4; ++j)
#pragma unroll
            for (int r = 0; r < 8; ++r) acc[i][j][r] = 0.0f;

    constexpr int NK = K_DIM / ABK;   // 32
    stage(0);

    const int half = lane >> 4;
    const int l15  = lane & 15;

    for (int kt = 0; kt < NK; ++kt) {
        wait_async0();        // this wave's tile-kt DMA done
        __syncthreads();      // all waves' tile-kt staged; tile kt-1 reads done
        if (kt + 1 < NK) stage(kt + 1);   // overlaps with compute below

        const char* sb = &lds[(kt & 1) * STG];
        const unsigned short* As = (const unsigned short*)sb;
        const unsigned short* Bs = (const unsigned short*)(sb + A_BYTES);

        FragU bfrag[4];
#pragma unroll
        for (int in = 0; in < 4; ++in) {
            // lane = column n; lanes 0-15: K {0..7,16..23}; lanes 16-31: K {8..15,24..31}
            const unsigned short* bp =
                &Bs[(wn * 64 + in * 16 + l15) * ABK];
            bfrag[in].u.lo = *(const u32x4*)(bp + half * 8);
            bfrag[in].u.hi = *(const u32x4*)(bp + 16 + half * 8);
        }
#pragma unroll
        for (int im = 0; im < 4; ++im) {
            // lanes 0-15: K {0..7,16..23}; lanes 16-31: K {8..15,24..31}
            FragU a;
            const unsigned short* ap = &As[(wm * 64 + im * 16 + l15) * ABK];
            a.u.lo = *(const u32x4*)(ap + half * 8);
            a.u.hi = *(const u32x4*)(ap + 16 + half * 8);
#pragma unroll
            for (int in = 0; in < 4; ++in)
                acc[im][in] = __builtin_amdgcn_wmma_f32_16x16x32_bf16(
                    false, a.v, false, bfrag[in].v,
                    (short)0, acc[im][in], false, false);
        }
    }

    const float scale = wscale[0];
#pragma unroll
    for (int im = 0; im < 4; ++im) {
        const int row_base = m0 + wm * 64 + im * 16 + (lane >> 4) * 8;
#pragma unroll
        for (int in = 0; in < 4; ++in) {
            const int col = n0 + wn * 64 + in * 16 + (lane & 15);
            const float bv = bias[col];
#pragma unroll
            for (int r = 0; r < 8; ++r) {
                out[(size_t)(row_base + r) * N_DIM + col] =
                    fmaf(acc[im][in][r], scale, bv);
            }
        }
    }
}

// =================== fallback GEMM (small/absent workspace) ================

static constexpr int BM = 128, BN = 128, BK = 32;

template <bool PRE_W>
__global__ __launch_bounds__(256)
void qlinear_gemm(const float* __restrict__ x,
                  const int8_t* __restrict__ w,
                  const unsigned short* __restrict__ wT,   // bf16 [N][K]
                  const float* __restrict__ wscale,
                  const float* __restrict__ bias,
                  float* __restrict__ out)
{
    __shared__ __align__(16) unsigned short As[BM * BK];
    __shared__ __align__(16) unsigned short Bs[BN * BK];

    const int tid  = threadIdx.x;
    const int lane = tid & 31;
    const int wid  = tid >> 5;
    const int wm   = wid & 3;
    const int wn   = wid >> 2;

    const int m0 = blockIdx.y * BM;
    const int n0 = blockIdx.x * BN;

    const int srow = tid >> 1;
    const int scol = (tid & 1) * 16;

    v8f acc[2][4];
#pragma unroll
    for (int i = 0; i < 2; ++i)
#pragma unroll
        for (int j = 0; j < 4; ++j)
#pragma unroll
            for (int r = 0; r < 8; ++r) acc[i][j][r] = 0.0f;

    for (int k0 = 0; k0 < K_DIM; k0 += BK) {
        {   // stage A: fp32 -> bf16 -> LDS
            const float* xp = x + (size_t)(m0 + srow) * K_DIM + k0 + scol;
            f32x4 a0 = *(const f32x4*)(xp + 0);
            f32x4 a1 = *(const f32x4*)(xp + 4);
            f32x4 a2 = *(const f32x4*)(xp + 8);
            f32x4 a3 = *(const f32x4*)(xp + 12);
            u32x4 p0 = { cvt2(a0.x, a0.y), cvt2(a0.z, a0.w),
                         cvt2(a1.x, a1.y), cvt2(a1.z, a1.w) };
            u32x4 p1 = { cvt2(a2.x, a2.y), cvt2(a2.z, a2.w),
                         cvt2(a3.x, a3.y), cvt2(a3.z, a3.w) };
            *(u32x4*)&As[srow * BK + scol + 0] = p0;
            *(u32x4*)&As[srow * BK + scol + 8] = p1;
        }
        if constexpr (PRE_W) {
            const unsigned short* bp =
                wT + (size_t)(n0 + srow) * K_DIM + k0 + scol;
            u32x4 q0 = *(const u32x4*)(bp + 0);
            u32x4 q1 = *(const u32x4*)(bp + 8);
            *(u32x4*)&Bs[srow * BK + scol + 0] = q0;
            *(u32x4*)&Bs[srow * BK + scol + 8] = q1;
        } else {
            const int bng = tid & 31;
            const int bkg = tid >> 5;
            const int8_t* wp =
                w + (size_t)(k0 + bkg * 4) * N_DIM + (n0 + bng * 4);
            unsigned short bt[4][4];
#pragma unroll
            for (int kk = 0; kk < 4; ++kk) {
                int packed;
                memcpy(&packed, wp + (size_t)kk * N_DIM, 4);
#pragma unroll
                for (int j = 0; j < 4; ++j)
                    bt[j][kk] = f2bf((float)(int8_t)(packed >> (8 * j)));
            }
#pragma unroll
            for (int j = 0; j < 4; ++j) {
                u32x2 q = { (unsigned)bt[j][0] | ((unsigned)bt[j][1] << 16),
                            (unsigned)bt[j][2] | ((unsigned)bt[j][3] << 16) };
                *(u32x2*)&Bs[(bng * 4 + j) * BK + bkg * 4] = q;
            }
        }

        __syncthreads();

        const int half = lane >> 4;
        const int l15  = lane & 15;
        FragU afrag[2];
#pragma unroll
        for (int im = 0; im < 2; ++im) {
            const unsigned short* ap = &As[(wm * 32 + im * 16 + l15) * BK];
            afrag[im].u.lo = *(const u32x4*)(ap + half * 8);
            afrag[im].u.hi = *(const u32x4*)(ap + 16 + half * 8);
        }
        FragU bfrag[4];
#pragma unroll
        for (int in = 0; in < 4; ++in) {
            const unsigned short* bp =
                &Bs[(wn * 64 + in * 16 + l15) * BK];
            bfrag[in].u.lo = *(const u32x4*)(bp + half * 8);
            bfrag[in].u.hi = *(const u32x4*)(bp + 16 + half * 8);
        }
#pragma unroll
        for (int im = 0; im < 2; ++im)
#pragma unroll
            for (int in = 0; in < 4; ++in)
                acc[im][in] = __builtin_amdgcn_wmma_f32_16x16x32_bf16(
                    false, afrag[im].v, false, bfrag[in].v,
                    (short)0, acc[im][in], false, false);

        __syncthreads();
    }

    const float scale = wscale[0];
#pragma unroll
    for (int im = 0; im < 2; ++im) {
        const int row_base = m0 + wm * 32 + im * 16 + (lane >> 4) * 8;
#pragma unroll
        for (int in = 0; in < 4; ++in) {
            const int col = n0 + wn * 64 + in * 16 + (lane & 15);
            const float bv = bias[col];
#pragma unroll
            for (int r = 0; r < 8; ++r)
                out[(size_t)(row_base + r) * N_DIM + col] =
                    fmaf(acc[im][in][r], scale, bv);
        }
    }
}

// =============================== launcher ==================================

extern "C" void kernel_launch(void* const* d_in, const int* in_sizes, int n_in,
                              void* d_out, int out_size, void* d_ws, size_t ws_size,
                              hipStream_t stream) {
    (void)in_sizes; (void)n_in; (void)out_size;
    const float*  x     = (const float*)d_in[0];
    const int8_t* w     = (const int8_t*)d_in[1];
    const float*  scale = (const float*)d_in[2];
    const float*  bias  = (const float*)d_in[3];
    float* out = (float*)d_out;

    const size_t wT_bytes = (size_t)N_DIM * K_DIM * sizeof(unsigned short); //  8 MB
    const size_t xb_bytes = (size_t)M_DIM * K_DIM * sizeof(unsigned short); // 64 MB

    unsigned short* wT = (unsigned short*)d_ws;
    unsigned short* xb = (unsigned short*)((char*)d_ws + wT_bytes);

    const bool pre_w = ws_size >= wT_bytes;
    const bool pre_x = ws_size >= wT_bytes + xb_bytes;

    dim3 block(256);
    if (pre_w)
        hipLaunchKernelGGL(convert_w_kernel, dim3(N_DIM / 256), block, 0,
                           stream, w, wT);
    if (pre_x)
        hipLaunchKernelGGL(convert_x_kernel,
                           dim3((unsigned)(((size_t)M_DIM * K_DIM) / (256 * 8))),
                           block, 0, stream, x, xb);

    if (pre_x && pre_w) {
        dim3 grid(N_DIM / ABN, M_DIM / ABM);  // (32, 128)
        hipLaunchKernelGGL(qlinear_gemm_async, grid, block, 0, stream,
                           xb, wT, scale, bias, out);
    } else if (pre_w) {
        dim3 grid(N_DIM / BN, M_DIM / BM);    // (32, 256)
        hipLaunchKernelGGL((qlinear_gemm<true>), grid, block, 0, stream,
                           x, w, wT, scale, bias, out);
    } else {
        dim3 grid(N_DIM / BN, M_DIM / BM);
        hipLaunchKernelGGL((qlinear_gemm<false>), grid, block, 0, stream,
                           x, w, wT, scale, bias, out);
    }
}
